// Autoformer_34102040330360
// MI455X (gfx1250) — matrix operand order repaired and checked by reading, not compiled, not run on hardware
//
#include <hip/hip_runtime.h>
#include <math.h>

// ---------------------------------------------------------------------------
// Autoformer encoder layer for MI455X (gfx1250, wave32, WMMA).
//  - All six GEMMs use v_wmma_f32_16x16x32_bf16 (bf16 in, fp32 accumulate).
//  - Epilogue variants (relu / residual) are template-instantiated so the
//    store path is branch-free (round-1 asm showed per-element s_cbranch).
//  - Next-K-slab global_prefetch_b8 via __builtin_prefetch overlaps HBM
//    latency with WMMA compute.
//  - Auto-correlation via packed complex FFT (1536 = 3 x 512 mixed radix,
//    Stockham in LDS) instead of an O(L^2 D) Gram matrix (77 GFLOP avoided).
//  - Deterministic: no float atomics; partial cross-spectra reduced in order.
// ---------------------------------------------------------------------------

#define Bb   32
#define Ll   1536
#define Dd   512
#define DFF  2048
#define TOPK 7
#define PI_F 3.14159265358979323846f

typedef __attribute__((ext_vector_type(16))) __bf16 v16bf;
typedef __attribute__((ext_vector_type(8)))  __bf16 v8bf;
typedef __attribute__((ext_vector_type(8)))  float  v8f;

__device__ __forceinline__ __bf16 f2bf(float f) {
  union { float f; unsigned u; } v; v.f = f;
  unsigned r = v.u + 0x7FFFu + ((v.u >> 16) & 1u);   // round-to-nearest-even
  unsigned short h = (unsigned short)(r >> 16);
  return __builtin_bit_cast(__bf16, h);
}

// ---------------------------------------------------------------------------
// Generic WMMA GEMM: C = act(A[MxK] * B[KxN] + bias) (+ resid), fp32 I/O.
// Block tile 128x128, 8 waves, each wave 32(M)x64(N) = 2x4 wmma tiles, K-slab 32.
// LDS: A tile row-major bf16, B tile stored transposed (Bt[n][k]) so each lane
// reads contiguous K. Row stride 40 elems (80B): 16B-aligned, bank-conflict-free.
// ---------------------------------------------------------------------------
#define LDSTR 40

template <int RELU, int RESID>
__global__ __launch_bounds__(256) void wmma_gemm(
    const float* __restrict__ A, const float* __restrict__ Bw,
    const float* __restrict__ bias, const float* __restrict__ resid,
    float* __restrict__ C, int M, int N, int K)
{
  __shared__ __align__(16) __bf16 lA[128 * LDSTR];
  __shared__ __align__(16) __bf16 lB[128 * LDSTR];

  const int tid    = threadIdx.x;
  const int lane   = tid & 31;
  const int wave   = tid >> 5;
  const int blockM = blockIdx.y * 128;
  const int blockN = blockIdx.x * 128;
  const int waveM  = (wave & 3) * 32;
  const int waveN  = (wave >> 2) * 64;
  const int lm     = lane & 15;
  const bool upper = lane >= 16;

  const v8f vzero = {0.f,0.f,0.f,0.f,0.f,0.f,0.f,0.f};
  v8f acc[2][4];
  for (int i = 0; i < 2; ++i)
    for (int j = 0; j < 4; ++j) acc[i][j] = vzero;

  for (int k0 = 0; k0 < K; k0 += 32) {
    // --- stage A tile 128x32 (fp32 -> bf16) ---
    {
      int r = tid >> 3;            // 0..31
      int c = (tid & 7) * 4;       // 0..28
      #pragma unroll
      for (int p = 0; p < 4; ++p) {
        int row = r + p * 32;
        const float4 f = *(const float4*)&A[(size_t)(blockM + row) * K + k0 + c];
        __bf16* dst = &lA[row * LDSTR + c];
        dst[0] = f2bf(f.x); dst[1] = f2bf(f.y); dst[2] = f2bf(f.z); dst[3] = f2bf(f.w);
      }
    }
    // --- stage B tile 32x128, transposed into lB[n][k] ---
    {
      int kk = tid >> 3;           // 0..31
      #pragma unroll
      for (int p = 0; p < 4; ++p) {
        int nc = (tid & 7) + p * 8;   // 0..31
        int n  = nc * 4;
        const float4 f = *(const float4*)&Bw[(size_t)(k0 + kk) * N + blockN + n];
        lB[(n + 0) * LDSTR + kk] = f2bf(f.x);
        lB[(n + 1) * LDSTR + kk] = f2bf(f.y);
        lB[(n + 2) * LDSTR + kk] = f2bf(f.z);
        lB[(n + 3) * LDSTR + kk] = f2bf(f.w);
      }
    }
    // --- speculative prefetch of next K-slab (global_prefetch_b8) ---
    if (k0 + 32 < K) {
      __builtin_prefetch(&A[(size_t)(blockM + (tid >> 1)) * K + k0 + 32 + (tid & 1) * 16], 0, 1);
      __builtin_prefetch(&Bw[(size_t)(k0 + 32 + (tid >> 3)) * N + blockN + (tid & 7) * 16], 0, 1);
    }
    __syncthreads();

    // --- fragments (ISA 16-bit A layout: lanes 0-15 K{0-7,16-23}, 16-31 K{8-15,24-31}) ---
    v16bf af[2], bfr[4];
    {
      int kc = upper ? 8 : 0;
      #pragma unroll
      for (int mt = 0; mt < 2; ++mt) {
        int row = waveM + mt * 16 + lm;
        v8bf lo = *(const v8bf*)&lA[row * LDSTR + kc];
        v8bf hi = *(const v8bf*)&lA[row * LDSTR + kc + 16];
        v16bf a;
        #pragma unroll
        for (int e = 0; e < 8; ++e) { a[e] = lo[e]; a[e + 8] = hi[e]; }
        af[mt] = a;
      }
      int kb = upper ? 8 : 0;      // B: lanes 0-15 K{0-7,16-23}, lanes 16-31 K{8-15,24-31}
      #pragma unroll
      for (int nt = 0; nt < 4; ++nt) {
        int n = waveN + nt * 16 + lm;
        v8bf lo = *(const v8bf*)&lB[n * LDSTR + kb];
        v8bf hi = *(const v8bf*)&lB[n * LDSTR + kb + 16];
        v16bf b;
        #pragma unroll
        for (int e = 0; e < 8; ++e) { b[e] = lo[e]; b[e + 8] = hi[e]; }
        bfr[nt] = b;
      }
    }
    #pragma unroll
    for (int mt = 0; mt < 2; ++mt)
      #pragma unroll
      for (int nt = 0; nt < 4; ++nt)
        acc[mt][nt] = __builtin_amdgcn_wmma_f32_16x16x32_bf16(
            false, af[mt], false, bfr[nt], (short)0, acc[mt][nt], false, false);
    __syncthreads();
  }

  // --- epilogue: C layout lane l vgpr r -> M = r (+8 upper half), N = l%16 ---
  #pragma unroll
  for (int mt = 0; mt < 2; ++mt)
    #pragma unroll
    for (int nt = 0; nt < 4; ++nt) {
      int col = blockN + waveN + nt * 16 + lm;
      float bval = bias[col];
      int row0 = blockM + waveM + mt * 16 + (upper ? 8 : 0);
      #pragma unroll
      for (int r = 0; r < 8; ++r) {
        size_t off = (size_t)(row0 + r) * N + col;
        float v = acc[mt][nt][r] + bval;
        if (RELU)  v = fmaxf(v, 0.0f);
        if (RESID) v += resid[off];
        C[off] = v;
      }
    }
}

// ---------------------------------------------------------------------------
// (B,L,D) -> (B,D,L) tiled transpose (coalesced FFT loads afterwards).
// ---------------------------------------------------------------------------
__global__ void transpose_kernel(const float* __restrict__ in, float* __restrict__ out)
{
  __shared__ float tile[32][33];
  int b = blockIdx.z, l0 = blockIdx.x * 32, d0 = blockIdx.y * 32;
  int tx = threadIdx.x, ty = threadIdx.y;
  for (int r = ty; r < 32; r += 8)
    tile[r][tx] = in[((size_t)b * Ll + l0 + r) * Dd + d0 + tx];
  __syncthreads();
  for (int r = ty; r < 32; r += 8)
    out[((size_t)b * Dd + d0 + r) * Ll + l0 + tx] = tile[tx][r];
}

// ---------------------------------------------------------------------------
// 1536-pt complex FFT in LDS: split n%3 -> three 512-pt Stockham (9 radix-2
// stages, natural order in/out) -> radix-3 twiddle combine. Result in AR/AI.
// ---------------------------------------------------------------------------
__device__ void fft1536(float* ZR, float* ZI, float* AR, float* AI,
                        float* BR, float* BI, int tid, int nthr)
{
  for (int i = tid; i < 1536; i += nthr) {
    int j = i & 511, m = i >> 9;
    AR[i] = ZR[3 * j + m]; AI[i] = ZI[3 * j + m];
  }
  __syncthreads();
  float *sR = AR, *sI = AI, *dR = BR, *dI = BI;
  int s = 1;
  for (int t = 0; t < 9; ++t) {
    int mh = 256 >> t;
    for (int i = tid; i < 768; i += nthr) {
      int mb = i >> 8, r = i & 255;
      int p = r >> t, q = r & (s - 1);
      int base = mb << 9;
      int ia = base + q + s * p;
      int ib = ia + s * mh;
      float ar = sR[ia], ai = sI[ia];
      float br = sR[ib], bi = sI[ib];
      float ang = -PI_F * (float)p / (float)mh;
      float wr, wi; __sincosf(ang, &wi, &wr);
      float ur = ar - br, ui = ai - bi;
      int oa = base + q + s * (2 * p);
      dR[oa] = ar + br;            dI[oa] = ai + bi;
      dR[oa + s] = ur * wr - ui * wi;  dI[oa + s] = ur * wi + ui * wr;
    }
    __syncthreads();
    float* tp;
    tp = sR; sR = dR; dR = tp;
    tp = sI; sI = dI; dI = tp;
    s <<= 1;
  }
  // after 9 swaps: data in sR/sI (phys B); combine writes dR/dI (phys A)
  for (int k = tid; k < 1536; k += nthr) {
    int km = k & 511;
    float ang = -2.0f * PI_F * (float)k / 1536.0f;
    float w1r, w1i; __sincosf(ang, &w1i, &w1r);
    float w2r = w1r * w1r - w1i * w1i, w2i = 2.0f * w1r * w1i;
    float y0r = sR[km],        y0i = sI[km];
    float y1r = sR[512 + km],  y1i = sI[512 + km];
    float y2r = sR[1024 + km], y2i = sI[1024 + km];
    dR[k] = y0r + w1r * y1r - w1i * y1i + w2r * y2r - w2i * y2i;
    dI[k] = y0i + w1r * y1i + w1i * y1r + w2r * y2i + w2i * y2r;
  }
  __syncthreads();
}

// ---------------------------------------------------------------------------
// Forward pass: per (b, 8-channel chunk): FFT of z = q + i*k (packed trick),
// unpack Qf/Kf, accumulate cross-spectrum Qf*conj(Kf); write partial to global.
// ---------------------------------------------------------------------------
__global__ __launch_bounds__(256) void fft_fwd_kernel(
    const float* __restrict__ Qt, const float* __restrict__ Kt,
    float* __restrict__ PR, float* __restrict__ PI_)
{
  __shared__ float ZR[Ll], ZI[Ll], AR[Ll], AI[Ll], BR[Ll], BI[Ll], ACR[Ll], ACI[Ll];
  int tid = threadIdx.x;
  int b   = blockIdx.y;
  int d0  = blockIdx.x * 8;
  for (int k = tid; k < Ll; k += 256) { ACR[k] = 0.f; ACI[k] = 0.f; }
  for (int c = 0; c < 8; ++c) {
    size_t base = ((size_t)b * Dd + (d0 + c)) * Ll;
    for (int k = tid; k < Ll; k += 256) { ZR[k] = Qt[base + k]; ZI[k] = Kt[base + k]; }
    __syncthreads();
    fft1536(ZR, ZI, AR, AI, BR, BI, tid, 256);
    for (int k = tid; k < Ll; k += 256) {
      float fr = AR[k], fi = AI[k];
      int nk = (Ll - k) & (Ll - 1);
      float gr = AR[nk], gi = -AI[nk];           // conj(F[N-k])
      float qr = 0.5f * (fr + gr), qi = 0.5f * (fi + gi);   // Qf
      float dr = fr - gr,  di = fi - gi;
      float kr = 0.5f * di, ki = -0.5f * dr;                // Kf = (F-G)/(2i)
      ACR[k] += qr * kr + qi * ki;               // Qf * conj(Kf)
      ACI[k] += qi * kr - qr * ki;
    }
    __syncthreads();
  }
  size_t pb = ((size_t)b * 64 + blockIdx.x) * Ll;
  for (int k = tid; k < Ll; k += 256) { PR[pb + k] = ACR[k]; PI_[pb + k] = ACI[k]; }
}

// ---------------------------------------------------------------------------
// Deterministic reduce of 64 partials per batch, then IFFT via conj-FFT-conj.
// mean_value[b,tau] = Re(IDFT(S))/D  (fold 1/N and 1/D together).
// ---------------------------------------------------------------------------
__global__ __launch_bounds__(256) void ifft_kernel(
    const float* __restrict__ PR, const float* __restrict__ PI_,
    float* __restrict__ mean)
{
  __shared__ float ZR[Ll], ZI[Ll], AR[Ll], AI[Ll], BR[Ll], BI[Ll];
  int tid = threadIdx.x, b = blockIdx.x;
  for (int k = tid; k < Ll; k += 256) {
    float sr = 0.f, si = 0.f;
    for (int xb = 0; xb < 64; ++xb) {
      size_t o = ((size_t)(b * 64 + xb)) * Ll + k;
      sr += PR[o]; si += PI_[o];
    }
    ZR[k] = sr; ZI[k] = -si;                     // conj
  }
  __syncthreads();
  fft1536(ZR, ZI, AR, AI, BR, BI, tid, 256);
  const float sc = 1.0f / (1536.0f * 512.0f);
  for (int k = tid; k < Ll; k += 256) mean[b * Ll + k] = AR[k] * sc;
}

// ---------------------------------------------------------------------------
// Top-7 lags of batch-summed mean_value + per-batch softmax weights.
// ---------------------------------------------------------------------------
__global__ __launch_bounds__(256) void topk_kernel(
    const float* __restrict__ mean, int* __restrict__ idxg, float* __restrict__ tcg)
{
  __shared__ float g[Ll];
  __shared__ float rv[256];
  __shared__ int   ri[256];
  __shared__ int   sIdx[TOPK];
  int tid = threadIdx.x;
  for (int k = tid; k < Ll; k += 256) {
    float s = 0.f;
    for (int b = 0; b < Bb; ++b) s += mean[b * Ll + k];
    g[k] = s;
  }
  __syncthreads();
  for (int it = 0; it < TOPK; ++it) {
    float bv = -3.4e38f; int bi = 0;
    for (int k = tid; k < Ll; k += 256)
      if (g[k] > bv) { bv = g[k]; bi = k; }
    rv[tid] = bv; ri[tid] = bi;
    __syncthreads();
    for (int off = 128; off > 0; off >>= 1) {
      if (tid < off && rv[tid + off] > rv[tid]) { rv[tid] = rv[tid + off]; ri[tid] = ri[tid + off]; }
      __syncthreads();
    }
    if (tid == 0) { sIdx[it] = ri[0]; idxg[it] = ri[0]; g[ri[0]] = -3.4e38f; }
    __syncthreads();
  }
  if (tid < Bb) {
    float w[TOPK]; float mx = -3.4e38f;
    for (int i = 0; i < TOPK; ++i) { w[i] = mean[tid * Ll + sIdx[i]]; mx = fmaxf(mx, w[i]); }
    float sum = 0.f;
    for (int i = 0; i < TOPK; ++i) { w[i] = __expf(w[i] - mx); sum += w[i]; }
    float inv = 1.0f / sum;
    for (int i = 0; i < TOPK; ++i) tcg[tid * TOPK + i] = w[i] * inv;
  }
}

// ---------------------------------------------------------------------------
// agg[b,t,:] = sum_i tc[b,i] * V[b,(t+idx[i])%L,:]   (float4, coalesced)
// ---------------------------------------------------------------------------
__global__ __launch_bounds__(256) void agg_kernel(
    const float* __restrict__ V, const int* __restrict__ idxg,
    const float* __restrict__ tcg, float* __restrict__ O)
{
  int gid = blockIdx.x * 256 + threadIdx.x;
  int d4  = gid & 127;
  int t   = (gid >> 7) % Ll;
  int b   = gid / (128 * Ll);
  size_t base = ((size_t)b * Ll) * Dd + (size_t)d4 * 4;
  float ox = 0.f, oy = 0.f, oz = 0.f, ow = 0.f;
  for (int i = 0; i < TOPK; ++i) {
    int tt = t + idxg[i]; if (tt >= Ll) tt -= Ll;
    const float4 v = *(const float4*)&V[base + (size_t)tt * Dd];
    float w = tcg[b * TOPK + i];
    ox += w * v.x; oy += w * v.y; oz += w * v.z; ow += w * v.w;
  }
  float4 o = make_float4(ox, oy, oz, ow);
  *(float4*)&O[base + (size_t)t * Dd] = o;
}

// ---------------------------------------------------------------------------
// y = x - moving_avg(x), k=25, edge-replicated == index-clamped 25-tap box.
// 25x reads mostly hit the 192MB L2 (whole tensor fits).
// ---------------------------------------------------------------------------
__global__ __launch_bounds__(256) void ma_sub_kernel(
    const float* __restrict__ X, float* __restrict__ Y)
{
  int gid = blockIdx.x * 256 + threadIdx.x;
  int d4  = gid & 127;
  int t   = (gid >> 7) % Ll;
  int b   = gid / (128 * Ll);
  size_t base = ((size_t)b * Ll) * Dd + (size_t)d4 * 4;
  float sx = 0.f, sy = 0.f, sz = 0.f, sw = 0.f;
  float4 ctr = make_float4(0.f, 0.f, 0.f, 0.f);
  for (int j = -12; j <= 12; ++j) {
    int tt = t + j; tt = tt < 0 ? 0 : (tt >= Ll ? Ll - 1 : tt);
    float4 v = *(const float4*)&X[base + (size_t)tt * Dd];
    sx += v.x; sy += v.y; sz += v.z; sw += v.w;
    if (j == 0) ctr = v;
  }
  const float inv = 1.0f / 25.0f;
  float4 o = make_float4(ctr.x - sx * inv, ctr.y - sy * inv,
                         ctr.z - sz * inv, ctr.w - sw * inv);
  *(float4*)&Y[base + (size_t)t * Dd] = o;
}

// ---------------------------------------------------------------------------
extern "C" void kernel_launch(void* const* d_in, const int* in_sizes, int n_in,
                              void* d_out, int out_size, void* d_ws, size_t ws_size,
                              hipStream_t stream) {
  (void)in_sizes; (void)n_in; (void)out_size; (void)ws_size;
  const float* x  = (const float*)d_in[0];
  const float* Wq = (const float*)d_in[1];
  const float* bq = (const float*)d_in[2];
  const float* Wk = (const float*)d_in[3];
  const float* bk = (const float*)d_in[4];
  const float* Wv = (const float*)d_in[5];
  const float* bv = (const float*)d_in[6];
  const float* Wo = (const float*)d_in[7];
  const float* bo = (const float*)d_in[8];
  const float* W1 = (const float*)d_in[9];
  const float* b1 = (const float*)d_in[10];
  const float* W2 = (const float*)d_in[11];
  const float* b2 = (const float*)d_in[12];
  float* out = (float*)d_out;
  char*  ws  = (char*)d_ws;

  const size_t SPLANE  = (size_t)Bb * 64 * Ll * 4;        // 12 MB partial plane
  float* PR   = (float*)(ws);
  float* PI_  = (float*)(ws + SPLANE);
  float* MEAN = (float*)(ws + 2 * SPLANE);
  int*   IDX  = (int*)  (ws + 2 * SPLANE + 256 * 1024);
  float* TC   = (float*)(ws + 2 * SPLANE + 256 * 1024 + 256);

  const size_t OFF_BIG = 32ull << 20;
  const size_t SZ = (size_t)Bb * Ll * Dd * 4;             // 96 MB
  float* Qb = (float*)(ws + OFF_BIG + 0 * SZ);
  float* Kb = (float*)(ws + OFF_BIG + 1 * SZ);
  float* Vb = (float*)(ws + OFF_BIG + 2 * SZ);
  float* Qt = (float*)(ws + OFF_BIG + 3 * SZ);
  float* Kt = (float*)(ws + OFF_BIG + 4 * SZ);
  float* Hb = (float*)(ws + OFF_BIG + 5 * SZ);            // 384 MB (DFF)
  // dead-buffer reuse (stream-ordered, no overlap of live ranges):
  float* AGG  = Qb;   // Q dead after transpose
  float* TMP1 = Kb;   // K dead after transpose
  float* X1   = Qt;   // Qt dead after fft_fwd
  float* Zb   = Kt;   // Kt dead after fft_fwd

  dim3 blk(256);
  dim3 gD(Dd / 128, (Bb * Ll) / 128);     // (4, 384)
  dim3 gF(DFF / 128, (Bb * Ll) / 128);    // (16, 384)
  const int M = Bb * Ll;
  const int nblk = (Bb * Ll * Dd / 4) / 256;

  // QKV projections
  wmma_gemm<0,0><<<gD, blk, 0, stream>>>(x, Wq, bq, nullptr, Qb, M, Dd, Dd);
  wmma_gemm<0,0><<<gD, blk, 0, stream>>>(x, Wk, bk, nullptr, Kb, M, Dd, Dd);
  wmma_gemm<0,0><<<gD, blk, 0, stream>>>(x, Wv, bv, nullptr, Vb, M, Dd, Dd);

  // transpose for coalesced FFT channel access
  dim3 tgrid(Ll / 32, Dd / 32, Bb), tblk(32, 8);
  transpose_kernel<<<tgrid, tblk, 0, stream>>>(Qb, Qt);
  transpose_kernel<<<tgrid, tblk, 0, stream>>>(Kb, Kt);

  // auto-correlation (FFT cross-spectrum -> IDFT -> top-k -> shifted agg)
  fft_fwd_kernel<<<dim3(Dd / 8, Bb), blk, 0, stream>>>(Qt, Kt, PR, PI_);
  ifft_kernel<<<dim3(Bb), blk, 0, stream>>>(PR, PI_, MEAN);
  topk_kernel<<<dim3(1), blk, 0, stream>>>(MEAN, IDX, TC);
  agg_kernel<<<dim3(nblk), blk, 0, stream>>>(Vb, IDX, TC, AGG);

  // out-proj + residual, series-decomp, FFN + residual, final decomp
  wmma_gemm<0,1><<<gD, blk, 0, stream>>>(AGG, Wo, bo, x, TMP1, M, Dd, Dd);
  ma_sub_kernel<<<dim3(nblk), blk, 0, stream>>>(TMP1, X1);
  wmma_gemm<1,0><<<gF, blk, 0, stream>>>(X1, W1, b1, nullptr, Hb, M, DFF, Dd);
  wmma_gemm<0,1><<<gD, blk, 0, stream>>>(Hb, W2, b2, X1, Zb, M, Dd, DFF);
  ma_sub_kernel<<<dim3(nblk), blk, 0, stream>>>(Zb, out);
}
